// LSTM_67156108640398
// MI455X (gfx1250) — matrix operand order repaired and checked by reading, not compiled, not run on hardware
//
#include <hip/hip_runtime.h>
#include <hip/hip_bf16.h>
#include <math.h>

// ---------------- problem constants ----------------
#define SEQ   4096
#define IN_F  256
#define HID   1024
#define OUT_F 256
#define G4H   (4 * HID)          // 4096 gate rows

// recurrence kernel config
#define NB    128                // persistent blocks
#define HPB   (HID / NB)         // hidden units per block = 8
#define RPB   (4 * HPB)          // W_hh rows per block   = 32
#define NW    8                  // waves per 256-thread block
#define RPW   (RPB / NW)         // rows per wave         = 4

typedef _Float16 v16h __attribute__((ext_vector_type(16)));
typedef _Float16 v8h  __attribute__((ext_vector_type(8)));
typedef float    v8f  __attribute__((ext_vector_type(8)));

// ---------------- fp32 -> f16 conversion (X and W_ih) ----------------
__global__ __launch_bounds__(256) void cvt_f16_kernel(
    const float* __restrict__ x,  // [SEQ*IN_F]
    const float* __restrict__ w,  // [G4H*IN_F]  (same element count)
    _Float16* __restrict__ xh,
    _Float16* __restrict__ wh)
{
    int i = blockIdx.x * 256 + threadIdx.x;     // 0 .. 1048575
    xh[i] = (_Float16)x[i];
    wh[i] = (_Float16)w[i];
}

// ---------------- init: zero h double-buffer + sync counter ----------------
__global__ __launch_bounds__(256) void init_kernel(float* __restrict__ hbuf,
                                                   unsigned* __restrict__ counter)
{
    int i = blockIdx.x * 256 + threadIdx.x;
    if (i < 2 * HID) hbuf[i] = 0.0f;
    if (i == 0) *counter = 0u;
}

// ---------------- WMMA GEMM: gx = X @ W_ih^T + (b_ih + b_hh) ----------------
// One 16x16 f32 tile per wave, K = 256 in 8 steps of v_wmma_f32_16x16x32_f16.
__global__ __launch_bounds__(256) void gemm_gx_kernel(
    const _Float16* __restrict__ Ah,   // [SEQ][IN_F] f16, row-major
    const _Float16* __restrict__ Wh,   // [G4H][IN_F] f16, row-major
    const float* __restrict__ b_ih,
    const float* __restrict__ b_hh,
    float* __restrict__ gx)            // [SEQ][G4H] f32
{
    const int lane = threadIdx.x & 31;
    const int wave = threadIdx.x >> 5;
    const int tile = blockIdx.x * NW + wave;   // 65536 tiles
    const int tm   = tile >> 8;                // 256 tiles along SEQ
    const int tn   = tile & 255;               // 256 tiles along gates
    const int half = lane >> 4;
    const int l15  = lane & 15;

    // A fragment: lane l (half=0) holds row M=l, K = 0..15 of each k-chunk
    //             lane l+16     holds row M=l, K = 16..31
    const _Float16* arow = Ah + (size_t)(tm * 16 + l15) * IN_F;
    // B fragment: lane n (half=0) holds col N=n (= W row), K = 0..15;
    //             lane n+16 K = 16..31 (contiguous along W_ih's K axis)
    const _Float16* brow = Wh + (size_t)(tn * 16 + l15) * IN_F;

    v8f acc = {};
#pragma unroll
    for (int k0 = 0; k0 < IN_F; k0 += 32) {
        v16h a = *(const v16h*)(arow + k0 + 16 * half);
        v16h b = *(const v16h*)(brow + k0 + 16 * half);
        acc = __builtin_amdgcn_wmma_f32_16x16x32_f16(
            /*neg_a=*/false, a, /*neg_b=*/false, b,
            /*c_mod=*/(short)0, acc, /*reuse_a=*/false, /*reuse_b=*/false);
    }

    const int col  = tn * 16 + l15;            // gate index
    const float bias = b_ih[col] + b_hh[col];
#pragma unroll
    for (int r = 0; r < 8; ++r) {
        int row = tm * 16 + r + 8 * half;      // C layout: VGPR r -> M = r + 8*half
        gx[(size_t)row * G4H + col] = acc[r] + bias;
    }
}

// ---------------- grid-wide barrier (monotonic counter) ----------------
__device__ __forceinline__ void grid_sync(unsigned* counter, unsigned target)
{
    __syncthreads();
    __threadfence();                            // release this block's h stores
    if (threadIdx.x == 0) {
        __hip_atomic_fetch_add(counter, 1u, __ATOMIC_ACQ_REL,
                               __HIP_MEMORY_SCOPE_AGENT);
        while (__hip_atomic_load(counter, __ATOMIC_ACQUIRE,
                                 __HIP_MEMORY_SCOPE_AGENT) < target) {
            __builtin_amdgcn_s_sleep(2);
        }
    }
    __syncthreads();
}

// ---------------- persistent LSTM recurrence ----------------
// Block b owns hidden units [b*HPB, b*HPB+HPB). c lives in LDS for all 4096 steps.
__global__ __launch_bounds__(256, 1) void lstm_recur_kernel(
    const float* __restrict__ gx,      // [SEQ][G4H]
    const float* __restrict__ Whh,     // [G4H][HID] fp32 (L2-resident, 16 MB)
    float* __restrict__ hbuf,          // [2][HID] double buffer
    unsigned* __restrict__ counter)
{
    __shared__ float hs[HID];          // 4 KB staged h_t
    __shared__ float glds[RPB];        // 32 gate pre-activations
    __shared__ float clds[HPB];        // persistent cell state

    const int tid  = threadIdx.x;
    const int b    = blockIdx.x;
    const int lane = tid & 31;
    const int wave = tid >> 5;

    if (tid < HPB) clds[tid] = 0.0f;

    for (int t = 0; t < SEQ; ++t) {
        // stage h_t into LDS (coalesced float4)
        const float* hin = hbuf + (size_t)(t & 1) * HID;
        ((float4*)hs)[tid] = ((const float4*)hin)[tid];
        __syncthreads();

        // each wave: RPW rows of W_hh, 32-lane strided dot + shfl reduction
#pragma unroll
        for (int rr = 0; rr < RPW; ++rr) {
            const int r    = wave * RPW + rr;          // local row 0..31
            const int u    = r & (HPB - 1);            // hidden unit within block
            const int q    = r / HPB;                  // gate: 0=i 1=f 2=g 3=o
            const int grow = q * HID + b * HPB + u;    // global W_hh row
            const float* wrow = Whh + (size_t)grow * HID;
            float sum = 0.0f;
#pragma unroll 8
            for (int j = 0; j < HID; j += 32)
                sum = fmaf(wrow[j + lane], hs[j + lane], sum);
#pragma unroll
            for (int off = 16; off > 0; off >>= 1)
                sum += __shfl_xor(sum, off, 32);
            if (lane == 0)
                glds[r] = sum + gx[(size_t)t * G4H + grow];
        }
        __syncthreads();

        // cell update for this block's HPB units (PyTorch gate order i,f,g,o)
        if (tid < HPB) {
            const int u = tid;
            const float gi = glds[u];
            const float gf = glds[HPB + u];
            const float gg = glds[2 * HPB + u];
            const float go = glds[3 * HPB + u];
            const float ig = 1.0f / (1.0f + __expf(-gi));
            const float fg = 1.0f / (1.0f + __expf(-gf));
            const float og = 1.0f / (1.0f + __expf(-go));
            const float c  = fg * clds[u] + ig * tanhf(gg);
            clds[u] = c;
            hbuf[(size_t)((t + 1) & 1) * HID + b * HPB + u] = og * tanhf(c);
        }

        grid_sync(counter, (unsigned)NB * (unsigned)(t + 1));
    }
}

// ---------------- output projection: out = h_T @ W_out^T + b_out ----------------
__global__ __launch_bounds__(256) void outproj_kernel(
    const float* __restrict__ h,       // h_T = hbuf[0] (SEQ even)
    const float* __restrict__ Wout,    // [OUT_F][HID]
    const float* __restrict__ bout,
    float* __restrict__ out)
{
    __shared__ float red[256];
    const int j   = blockIdx.x;
    const int tid = threadIdx.x;
    const float* wrow = Wout + (size_t)j * HID;
    float s = 0.0f;
    for (int k = tid; k < HID; k += 256)
        s = fmaf(wrow[k], h[k], s);
    red[tid] = s;
    __syncthreads();
    for (int w = 128; w > 0; w >>= 1) {
        if (tid < w) red[tid] += red[tid + w];
        __syncthreads();
    }
    if (tid == 0) out[j] = red[0] + bout[j];
}

// ---------------- host launcher ----------------
extern "C" void kernel_launch(void* const* d_in, const int* in_sizes, int n_in,
                              void* d_out, int out_size, void* d_ws, size_t ws_size,
                              hipStream_t stream)
{
    (void)in_sizes; (void)n_in; (void)out_size; (void)ws_size;

    const float* input_seq = (const float*)d_in[0];   // [SEQ, IN_F]
    const float* W_ih      = (const float*)d_in[1];   // [G4H, IN_F]
    const float* W_hh      = (const float*)d_in[2];   // [G4H, HID]
    const float* b_ih      = (const float*)d_in[3];   // [G4H]
    const float* b_hh      = (const float*)d_in[4];   // [G4H]
    const float* W_out     = (const float*)d_in[5];   // [OUT_F, HID]
    const float* b_out     = (const float*)d_in[6];   // [OUT_F]
    float*       out       = (float*)d_out;           // [OUT_F]

    // workspace layout (all offsets 32B+ aligned)
    char* ws = (char*)d_ws;
    _Float16* Ah   = (_Float16*)(ws);                               // 2 MB
    _Float16* Wh   = (_Float16*)(ws + (size_t)2  * 1024 * 1024);    // 2 MB
    float*    gx   = (float*)   (ws + (size_t)4  * 1024 * 1024);    // 64 MB
    float*    hbuf = (float*)   (ws + (size_t)68 * 1024 * 1024);    // 8 KB
    unsigned* ctr  = (unsigned*)(ws + (size_t)68 * 1024 * 1024 + 8192);

    // 1) fp32 -> f16 for X and W_ih (1M elements each)
    cvt_f16_kernel<<<(SEQ * IN_F) / 256, 256, 0, stream>>>(input_seq, W_ih, Ah, Wh);

    // 2) zero h double-buffer + reset grid-sync counter (graph-replay safe)
    init_kernel<<<9, 256, 0, stream>>>(hbuf, ctr);

    // 3) WMMA GEMM: gx_all = X @ W_ih^T + (b_ih + b_hh)
    //    65536 16x16 tiles, 8 waves/block -> 8192 blocks
    gemm_gx_kernel<<<(SEQ / 16) * (G4H / 16) / NW, 256, 0, stream>>>(
        Ah, Wh, b_ih, b_hh, gx);

    // 4) persistent recurrence: 128 blocks, grid-synced per timestep
    lstm_recur_kernel<<<NB, 256, 0, stream>>>(gx, W_hh, hbuf, ctr);

    // 5) output projection from h_T (= hbuf[0], since SEQ is even)
    outproj_kernel<<<OUT_F, 256, 0, stream>>>(hbuf, W_out, b_out, out);
}
